// MagnusChessTransformer_56246891708886
// MI455X (gfx1250) — matrix operand order repaired and checked by reading, not compiled, not run on hardware
//
#include <hip/hip_runtime.h>
#include <hip/hip_bf16.h>
#include <math.h>

// ---------------- CDNA5 WMMA types ----------------
typedef __bf16 bf16;
typedef __attribute__((ext_vector_type(16))) __bf16 v16bf;
typedef __attribute__((ext_vector_type(4)))  __bf16 v4bf;
typedef __attribute__((ext_vector_type(8)))  float  v8f;

#define B_   128
#define S_   64
#define D_   512
#define H_   8
#define HD_  64
#define FF_  2048
#define L_   12
#define MTOK (B_ * S_)   // 8192 tokens

__device__ __forceinline__ float gelu_exact(float x) {
  return 0.5f * x * (1.0f + erff(x * 0.70710678118654752f));
}

__device__ __forceinline__ v8f v8f_zero() {
  v8f z;
#pragma unroll
  for (int e = 0; e < 8; ++e) z[e] = 0.0f;
  return z;
}

// pack 4 floats -> 4 bf16, single 8-byte LDS store (ds_store_b64)
__device__ __forceinline__ void st4bf(bf16* dst, float a, float b, float c, float d) {
  v4bf w;
  w[0] = (bf16)a; w[1] = (bf16)b; w[2] = (bf16)c; w[3] = (bf16)d;
  *(v4bf*)dst = w;
}

// A fragment, 16x32 bf16 (ISA 7.12.2): lane<16 -> row=lane, K={kb..kb+7, kb+16..kb+23}, kb=0;
// lane>=16 -> kb=8. Row-major source, leading dim `ld`.
__device__ __forceinline__ v16bf ldfragA_bf16(const bf16* base, int ld, int lane) {
  const int r  = lane & 15;
  const int kb = (lane >> 4) << 3;
  const bf16* p = base + r * ld + kb;
  v16bf f;
#pragma unroll
  for (int e = 0; e < 8; ++e) f[e] = p[e];
#pragma unroll
  for (int e = 0; e < 8; ++e) f[8 + e] = p[16 + e];
  return f;
}

// Same fragment converting from fp32 source (softmax probs).
__device__ __forceinline__ v16bf ldfragA_f32(const float* base, int ld, int lane) {
  const int r  = lane & 15;
  const int kb = (lane >> 4) << 3;
  const float* p = base + r * ld + kb;
  v16bf f;
#pragma unroll
  for (int e = 0; e < 8; ++e) f[e] = (bf16)p[e];
#pragma unroll
  for (int e = 0; e < 8; ++e) f[8 + e] = (bf16)p[16 + e];
  return f;
}

// B fragment, 32x16 bf16, from transposed LDS image Bt[n][k]:
// lane<16 -> col=lane, K={0..7, 16..23}; lane>=16 -> col=lane-16, K={8..15, 24..31} (matches A).
__device__ __forceinline__ v16bf ldfragB_bf16(const bf16* baseT, int ld, int lane) {
  const int n  = lane & 15;
  const int kb = (lane >> 4) << 3;
  const bf16* p = baseT + n * ld + kb;
  v16bf f;
#pragma unroll
  for (int e = 0; e < 8; ++e) f[e] = p[e];
#pragma unroll
  for (int e = 0; e < 8; ++e) f[8 + e] = p[16 + e];
  return f;
}

// ---------------- WMMA GEMM, 128x128 block tile ----------------
// 256 threads = 8 waves; wave w -> rows 16w..16w+15, 8 col-tiles (A frag reused 8x).
__global__ __launch_bounds__(256) void wmma_gemm_bias_128(
    const float* __restrict__ A, const float* __restrict__ W,
    const float* __restrict__ bias, float* __restrict__ C,
    int M, int N, int K) {
  __shared__ bf16 As[128 * 32];   // A tile, row-major
  __shared__ bf16 Bt[128 * 32];   // B tile, transposed: Bt[n][k]
  const int tid  = threadIdx.x;
  const int wave = tid >> 5, lane = tid & 31;
  const int row0 = blockIdx.x * 128;
  const int col0 = blockIdx.y * 128;
  (void)M;
  v8f acc[8];
#pragma unroll
  for (int j = 0; j < 8; ++j) acc[j] = v8f_zero();

  const int bn    = tid & 127;          // B staging column
  const int bhalf = tid >> 7;           // 0 or 1

  for (int k0 = 0; k0 < K; k0 += 32) {
    // ---- batched global loads (b128) into registers ----
    float4 av[4];
#pragma unroll
    for (int p = 0; p < 4; ++p) {
      const int i = tid + p * 256;      // 0..1023
      const int r = i >> 3, cq = i & 7;
      av[p] = *(const float4*)&A[(size_t)(row0 + r) * K + (k0 + 4 * cq)];
    }
    float bvv[4][4];
#pragma unroll
    for (int p = 0; p < 4; ++p) {
      const int kb4 = 4 * (bhalf + 2 * p);
#pragma unroll
      for (int kk = 0; kk < 4; ++kk)
        bvv[p][kk] = W[(size_t)(k0 + kb4 + kk) * N + (col0 + bn)];
    }
    if (k0 + 32 < K) {  // global_prefetch_b8 for next K slab
      __builtin_prefetch(&A[(size_t)(row0 + (tid >> 1)) * K + (k0 + 32)], 0, 1);
      __builtin_prefetch(&W[(size_t)(k0 + 32 + (tid >> 3)) * N + col0 + ((tid & 7) << 4)], 0, 1);
    }
    __syncthreads();
    // ---- convert + packed LDS stores ----
#pragma unroll
    for (int p = 0; p < 4; ++p) {
      const int i = tid + p * 256;
      const int r = i >> 3, cq = i & 7;
      st4bf(&As[r * 32 + 4 * cq], av[p].x, av[p].y, av[p].z, av[p].w);
    }
#pragma unroll
    for (int p = 0; p < 4; ++p) {
      const int kb4 = 4 * (bhalf + 2 * p);
      st4bf(&Bt[bn * 32 + kb4], bvv[p][0], bvv[p][1], bvv[p][2], bvv[p][3]);
    }
    __syncthreads();
    const v16bf a = ldfragA_bf16(As + (wave * 16) * 32, 32, lane);
#pragma unroll
    for (int j = 0; j < 8; ++j) {
      const v16bf bfrag = ldfragB_bf16(Bt + (j * 16) * 32, 32, lane);
      acc[j] = __builtin_amdgcn_wmma_f32_16x16x32_bf16(
          false, a, false, bfrag, (short)0, acc[j], false, false);
    }
  }

  const int rbase = (lane >> 4) << 3;
  const int cc    = lane & 15;
#pragma unroll
  for (int j = 0; j < 8; ++j) {
    const int col = col0 + j * 16 + cc;
    const float bv = (bias != nullptr) ? bias[col] : 0.0f;
#pragma unroll
    for (int e = 0; e < 8; ++e) {
      const int row = row0 + wave * 16 + rbase + e;
      C[(size_t)row * N + col] = acc[j][e] + bv;
    }
  }
}

// ---------------- WMMA GEMM, 64x64 block tile (for N % 128 != 0) ----------------
__global__ __launch_bounds__(128) void wmma_gemm_bias_64(
    const float* __restrict__ A, const float* __restrict__ W,
    const float* __restrict__ bias, float* __restrict__ C,
    int M, int N, int K) {
  __shared__ bf16 As[64 * 32];
  __shared__ bf16 Bt[64 * 32];
  const int tid  = threadIdx.x;
  const int wave = tid >> 5, lane = tid & 31;
  const int row0 = blockIdx.x * 64;
  const int col0 = blockIdx.y * 64;
  (void)M;
  v8f acc[4];
#pragma unroll
  for (int j = 0; j < 4; ++j) acc[j] = v8f_zero();

  const int bn    = tid & 63;
  const int bhalf = tid >> 6;

  for (int k0 = 0; k0 < K; k0 += 32) {
    float4 av[4];
#pragma unroll
    for (int p = 0; p < 4; ++p) {
      const int i = tid + p * 128;      // 0..511
      const int r = i >> 3, cq = i & 7;
      av[p] = *(const float4*)&A[(size_t)(row0 + r) * K + (k0 + 4 * cq)];
    }
    float bvv[4][4];
#pragma unroll
    for (int p = 0; p < 4; ++p) {
      const int kb4 = 4 * (bhalf + 2 * p);
#pragma unroll
      for (int kk = 0; kk < 4; ++kk)
        bvv[p][kk] = W[(size_t)(k0 + kb4 + kk) * N + (col0 + bn)];
    }
    if (k0 + 32 < K) {
      __builtin_prefetch(&A[(size_t)(row0 + (tid >> 1)) * K + (k0 + 32)], 0, 1);
      __builtin_prefetch(&W[(size_t)(k0 + 32 + (tid >> 2)) * N + col0 + ((tid & 3) << 4)], 0, 1);
    }
    __syncthreads();
#pragma unroll
    for (int p = 0; p < 4; ++p) {
      const int i = tid + p * 128;
      const int r = i >> 3, cq = i & 7;
      st4bf(&As[r * 32 + 4 * cq], av[p].x, av[p].y, av[p].z, av[p].w);
    }
#pragma unroll
    for (int p = 0; p < 4; ++p) {
      const int kb4 = 4 * (bhalf + 2 * p);
      st4bf(&Bt[bn * 32 + kb4], bvv[p][0], bvv[p][1], bvv[p][2], bvv[p][3]);
    }
    __syncthreads();
    const v16bf a = ldfragA_bf16(As + (wave * 16) * 32, 32, lane);
#pragma unroll
    for (int j = 0; j < 4; ++j) {
      const v16bf bfrag = ldfragB_bf16(Bt + (j * 16) * 32, 32, lane);
      acc[j] = __builtin_amdgcn_wmma_f32_16x16x32_bf16(
          false, a, false, bfrag, (short)0, acc[j], false, false);
    }
  }

  const int rbase = (lane >> 4) << 3;
  const int cc    = lane & 15;
#pragma unroll
  for (int j = 0; j < 4; ++j) {
    const int col = col0 + j * 16 + cc;
    const float bv = (bias != nullptr) ? bias[col] : 0.0f;
#pragma unroll
    for (int e = 0; e < 8; ++e) {
      const int row = row0 + wave * 16 + rbase + e;
      C[(size_t)row * N + col] = acc[j][e] + bv;
    }
  }
}

// ---------------- Fused attention per (batch, head) ----------------
// E = Q @ rel^T (WMMA), scores = (Q K^T)*WMMA + E-gather, softmax fp32, ctx = P V (WMMA).
__global__ __launch_bounds__(256) void attention_kernel(
    const float* __restrict__ Q, const float* __restrict__ Km,
    const float* __restrict__ V, const float* __restrict__ relW,
    float* __restrict__ ctx) {
  __shared__ bf16  Qb[S_ * HD_];    // [q][d]
  __shared__ bf16  Kb[S_ * HD_];    // [key][d]  == Bt for QK^T
  __shared__ bf16  Vt[HD_ * S_];    // [d][key]  == Bt for P*V
  __shared__ bf16  Rb[128 * HD_];   // rel embeddings [ridx][d], row 127 zero
  __shared__ float Sc[S_ * S_];     // scores / probs (fp32)
  __shared__ float E[S_ * 128];     // rel-bias matrix Q@rel^T (fp32)
  const int tid = threadIdx.x;
  const int wave = tid >> 5, lane = tid & 31;
  const int bi = blockIdx.x >> 3, h = blockIdx.x & 7;
  const size_t gbase = ((size_t)bi * S_) * D_ + (size_t)h * HD_;

  // ---- batched vector staging of Q, K, V^T ----
  {
    float4 qa[4], ka[4], va[4];
#pragma unroll
    for (int p = 0; p < 4; ++p) {
      const int i = tid + p * 256;            // 0..1023
      const int s = i >> 4, dq = i & 15;
      const size_t off = gbase + (size_t)s * D_ + 4 * dq;
      qa[p] = *(const float4*)(Q  + off);
      ka[p] = *(const float4*)(Km + off);
      va[p] = *(const float4*)(V  + off);
    }
#pragma unroll
    for (int p = 0; p < 4; ++p) {
      const int i = tid + p * 256;
      const int s = i >> 4, dq = i & 15;
      st4bf(&Qb[s * 64 + 4 * dq], qa[p].x, qa[p].y, qa[p].z, qa[p].w);
      st4bf(&Kb[s * 64 + 4 * dq], ka[p].x, ka[p].y, ka[p].z, ka[p].w);
      Vt[(4 * dq + 0) * 64 + s] = (bf16)va[p].x;
      Vt[(4 * dq + 1) * 64 + s] = (bf16)va[p].y;
      Vt[(4 * dq + 2) * 64 + s] = (bf16)va[p].z;
      Vt[(4 * dq + 3) * 64 + s] = (bf16)va[p].w;
    }
#pragma unroll
    for (int p = 0; p < 8; ++p) {
      const int i = tid + p * 256;            // 0..2047
      const int r = i >> 4, dq = i & 15;
      if (r < 127) {
        const float4 rr = *(const float4*)(relW + (size_t)r * HD_ + 4 * dq);
        st4bf(&Rb[r * 64 + 4 * dq], rr.x, rr.y, rr.z, rr.w);
      } else {
        st4bf(&Rb[r * 64 + 4 * dq], 0.0f, 0.0f, 0.0f, 0.0f);
      }
    }
  }
  __syncthreads();

  const int rbase = (lane >> 4) << 3;
  const int cc    = lane & 15;

  // ---- E = Q(64x64) @ rel^T(64x128): 32 tiles, 4 per wave ----
#pragma unroll
  for (int ti = 0; ti < 4; ++ti) {
    const int t = wave * 4 + ti;
    const int tr = t >> 3, tc = t & 7;
    v8f acc = v8f_zero();
#pragma unroll
    for (int ks = 0; ks < 2; ++ks) {
      const v16bf a = ldfragA_bf16(Qb + tr * 16 * HD_ + ks * 32, HD_, lane);
      const v16bf b = ldfragB_bf16(Rb + tc * 16 * HD_ + ks * 32, HD_, lane);
      acc = __builtin_amdgcn_wmma_f32_16x16x32_bf16(false, a, false, b, (short)0, acc, false, false);
    }
#pragma unroll
    for (int e = 0; e < 8; ++e)
      E[(tr * 16 + rbase + e) * 128 + tc * 16 + cc] = acc[e];
  }
  __syncthreads();

  // ---- scores = (Q K^T + E-gather) * 1/sqrt(64): 16 tiles, 2 per wave ----
#pragma unroll
  for (int ti = 0; ti < 2; ++ti) {
    const int t = wave * 2 + ti;
    const int tr = t >> 2, tc = t & 3;
    v8f acc = v8f_zero();
#pragma unroll
    for (int ks = 0; ks < 2; ++ks) {
      const v16bf a = ldfragA_bf16(Qb + tr * 16 * HD_ + ks * 32, HD_, lane);
      const v16bf b = ldfragB_bf16(Kb + tc * 16 * HD_ + ks * 32, HD_, lane);
      acc = __builtin_amdgcn_wmma_f32_16x16x32_bf16(false, a, false, b, (short)0, acc, false, false);
    }
#pragma unroll
    for (int e = 0; e < 8; ++e) {
      const int qi = tr * 16 + rbase + e;
      const int ki = tc * 16 + cc;
      const int ridx = ki - qi + 63;          // in [0,126]
      Sc[qi * S_ + ki] = (acc[e] + E[qi * 128 + ridx]) * 0.125f;
    }
  }
  __syncthreads();

  // ---- softmax, one row per thread (fp32) ----
  if (tid < S_) {
    float mx = -3.0e38f;
    for (int kk = 0; kk < S_; ++kk) mx = fmaxf(mx, Sc[tid * S_ + kk]);
    float sum = 0.0f;
    for (int kk = 0; kk < S_; ++kk) {
      const float e2 = expf(Sc[tid * S_ + kk] - mx);
      Sc[tid * S_ + kk] = e2;
      sum += e2;
    }
    const float inv = 1.0f / sum;
    for (int kk = 0; kk < S_; ++kk) Sc[tid * S_ + kk] *= inv;
  }
  __syncthreads();

  // ---- ctx = P @ V ----
#pragma unroll
  for (int ti = 0; ti < 2; ++ti) {
    const int t = wave * 2 + ti;
    const int tr = t >> 2, tc = t & 3;
    v8f acc = v8f_zero();
#pragma unroll
    for (int ks = 0; ks < 2; ++ks) {
      const v16bf a = ldfragA_f32(Sc + tr * 16 * S_ + ks * 32, S_, lane);
      const v16bf b = ldfragB_bf16(Vt + tc * 16 * S_ + ks * 32, S_, lane);
      acc = __builtin_amdgcn_wmma_f32_16x16x32_bf16(false, a, false, b, (short)0, acc, false, false);
    }
#pragma unroll
    for (int e = 0; e < 8; ++e) {
      const int qi = tr * 16 + rbase + e;
      const int di = tc * 16 + cc;
      ctx[gbase + (size_t)qi * D_ + di] = acc[e];
    }
  }
}

// ---------------- Conv (3x3 or 1x1, SAME, 8x8) + BN(eval) + GELU ----------------
__global__ __launch_bounds__(64) void conv_kernel(
    const float* __restrict__ in, const float* __restrict__ w,
    const float* __restrict__ bias, const float* __restrict__ g,
    const float* __restrict__ beta, float* __restrict__ out,
    int Cin, int Cout, int ks) {
  __shared__ float wsm[2304];              // up to 256*3*3
  const int co = blockIdx.x % Cout;
  const int bi = blockIdx.x / Cout;
  const int nw = Cin * ks * ks;
  for (int i = threadIdx.x; i < nw; i += 64) wsm[i] = w[(size_t)co * nw + i];
  __syncthreads();
  const int y = threadIdx.x >> 3, x = threadIdx.x & 7;
  const int pad = ks >> 1;
  float s = 0.0f;
  for (int ci = 0; ci < Cin; ++ci) {
    const float* ip = in + ((size_t)bi * Cin + ci) * 64;
    const float* wp = wsm + ci * ks * ks;
    for (int ky = 0; ky < ks; ++ky) {
      const int yy = y + ky - pad;
      if (yy < 0 || yy > 7) continue;
      for (int kx = 0; kx < ks; ++kx) {
        const int xx = x + kx - pad;
        if (xx < 0 || xx > 7) continue;
        s += ip[yy * 8 + xx] * wp[ky * ks + kx];
      }
    }
  }
  const float bnscale = 0.99999500003749978f;  // 1/sqrt(1+1e-5)
  const float val = (s + bias[co]) * (bnscale * g[co]) + beta[co];
  out[((size_t)bi * Cout + co) * 64 + y * 8 + x] = gelu_exact(val);
}

// tokens[b,s,d] = conv[b,d,s] + pe[s,d] + learned_pe[s,d]
__global__ __launch_bounds__(256) void tokens_kernel(
    const float* __restrict__ conv, const float* __restrict__ pe,
    const float* __restrict__ lpe, float* __restrict__ t) {
  const int idx = blockIdx.x * 256 + threadIdx.x;
  if (idx >= MTOK * D_) return;
  const int d = idx & 511;
  const int s = (idx >> 9) & 63;
  const int bi = idx >> 15;
  t[idx] = conv[((size_t)bi * D_ + d) * 64 + s] + pe[s * D_ + d] + lpe[s * D_ + d];
}

// t = LN(t + delta) over last dim 512 (delta may be null). One block per row.
__global__ __launch_bounds__(256) void add_ln_kernel(
    float* __restrict__ t, const float* __restrict__ delta,
    const float* __restrict__ g, const float* __restrict__ b) {
  __shared__ float red[256];
  const int row = blockIdx.x, tid = threadIdx.x;
  float v0 = t[(size_t)row * 512 + tid];
  float v1 = t[(size_t)row * 512 + tid + 256];
  if (delta != nullptr) {
    v0 += delta[(size_t)row * 512 + tid];
    v1 += delta[(size_t)row * 512 + tid + 256];
  }
  red[tid] = v0 + v1;
  __syncthreads();
  for (int o = 128; o > 0; o >>= 1) { if (tid < o) red[tid] += red[tid + o]; __syncthreads(); }
  const float mu = red[0] * (1.0f / 512.0f);
  __syncthreads();
  const float d0 = v0 - mu, d1 = v1 - mu;
  red[tid] = d0 * d0 + d1 * d1;
  __syncthreads();
  for (int o = 128; o > 0; o >>= 1) { if (tid < o) red[tid] += red[tid + o]; __syncthreads(); }
  const float rs = rsqrtf(red[0] * (1.0f / 512.0f) + 1e-5f);
  t[(size_t)row * 512 + tid]       = d0 * rs * g[tid]       + b[tid];
  t[(size_t)row * 512 + tid + 256] = d1 * rs * g[tid + 256] + b[tid + 256];
}

// In-place LN over last dim N (<=2048), optional exact GELU. One block per row.
__global__ __launch_bounds__(256) void ln_act_kernel(
    float* __restrict__ X, const float* __restrict__ g,
    const float* __restrict__ b, int N, int act) {
  __shared__ float red[256];
  const int row = blockIdx.x, tid = threadIdx.x;
  float vals[8];
  float s = 0.0f;
  int cnt = 0;
  for (int i = tid; i < N; i += 256) { const float v = X[(size_t)row * N + i]; vals[cnt++] = v; s += v; }
  red[tid] = s; __syncthreads();
  for (int o = 128; o > 0; o >>= 1) { if (tid < o) red[tid] += red[tid + o]; __syncthreads(); }
  const float mu = red[0] / (float)N;
  __syncthreads();
  s = 0.0f;
  for (int c = 0; c < cnt; ++c) { const float d = vals[c] - mu; s += d * d; }
  red[tid] = s; __syncthreads();
  for (int o = 128; o > 0; o >>= 1) { if (tid < o) red[tid] += red[tid + o]; __syncthreads(); }
  const float rs = rsqrtf(red[0] / (float)N + 1e-5f);
  cnt = 0;
  for (int i = tid; i < N; i += 256) {
    float y = (vals[cnt++] - mu) * rs * g[i] + b[i];
    if (act) y = gelu_exact(y);
    X[(size_t)row * N + i] = y;
  }
}

__global__ __launch_bounds__(512) void mean_pool_kernel(
    const float* __restrict__ t, float* __restrict__ rep) {
  const int bi = blockIdx.x, d = threadIdx.x;
  float s = 0.0f;
  for (int ss = 0; ss < S_; ++ss) s += t[((size_t)bi * S_ + ss) * D_ + d];
  rep[(size_t)bi * D_ + d] = s * (1.0f / 64.0f);
}

__global__ __launch_bounds__(128) void value_final_kernel(
    const float* __restrict__ hh, const float* __restrict__ w,
    const float* __restrict__ b, float* __restrict__ out) {
  const int bi = threadIdx.x;
  float s = 0.0f;
  for (int j = 0; j < 128; ++j) s += hh[bi * 128 + j] * w[j];
  out[bi] = tanhf(s + b[0]);
}

// ---------------- host-side orchestration ----------------
static inline void launch_gemm(const float* A, const float* W, const float* bias,
                               float* C, int M, int N, int K, hipStream_t s) {
  if ((M % 128 == 0) && (N % 128 == 0)) {
    dim3 grid(M / 128, N / 128);
    wmma_gemm_bias_128<<<grid, 256, 0, s>>>(A, W, bias, C, M, N, K);
  } else {
    dim3 grid(M / 64, N / 64);
    wmma_gemm_bias_64<<<grid, 128, 0, s>>>(A, W, bias, C, M, N, K);
  }
}

extern "C" void kernel_launch(void* const* d_in, const int* in_sizes, int n_in,
                              void* d_out, int out_size, void* d_ws, size_t ws_size,
                              hipStream_t stream) {
  (void)in_sizes; (void)n_in; (void)out_size; (void)ws_size;
  const float* x = (const float*)d_in[0];
  const float *cw[4], *cb[4], *cg[4], *cbe[4];
  for (int i = 0; i < 4; ++i) {
    cw[i]  = (const float*)d_in[1 + 4 * i];
    cb[i]  = (const float*)d_in[2 + 4 * i];
    cg[i]  = (const float*)d_in[3 + 4 * i];
    cbe[i] = (const float*)d_in[4 + 4 * i];
  }
  const float* pe   = (const float*)d_in[17];
  const float* lpe  = (const float*)d_in[18];
  const float* Wq   = (const float*)d_in[19];
  const float* bq   = (const float*)d_in[20];
  const float* Wk   = (const float*)d_in[21];
  const float* bk   = (const float*)d_in[22];
  const float* Wv   = (const float*)d_in[23];
  const float* bv   = (const float*)d_in[24];
  const float* Wo   = (const float*)d_in[25];
  const float* bo   = (const float*)d_in[26];
  const float* relW = (const float*)d_in[27];
  const float* ln1g = (const float*)d_in[28];
  const float* ln1b = (const float*)d_in[29];
  const float* ln2g = (const float*)d_in[30];
  const float* ln2b = (const float*)d_in[31];
  const float* W1   = (const float*)d_in[32];
  const float* b1   = (const float*)d_in[33];
  const float* lnfg = (const float*)d_in[34];
  const float* lnfb = (const float*)d_in[35];
  const float* W2   = (const float*)d_in[36];
  const float* b2   = (const float*)d_in[37];
  const float* nfg  = (const float*)d_in[38];
  const float* nfb  = (const float*)d_in[39];

  const float *Pw[4], *Pb[4], *Pg[3], *Pbe[3];
  Pw[0]=(const float*)d_in[40]; Pb[0]=(const float*)d_in[41]; Pg[0]=(const float*)d_in[42]; Pbe[0]=(const float*)d_in[43];
  Pw[1]=(const float*)d_in[44]; Pb[1]=(const float*)d_in[45]; Pg[1]=(const float*)d_in[46]; Pbe[1]=(const float*)d_in[47];
  Pw[2]=(const float*)d_in[48]; Pb[2]=(const float*)d_in[49]; Pg[2]=(const float*)d_in[50]; Pbe[2]=(const float*)d_in[51];
  Pw[3]=(const float*)d_in[52]; Pb[3]=(const float*)d_in[53];

  const float *Vw[5], *Vb[5], *Vg[4], *Vbe[4];
  Vw[0]=(const float*)d_in[54]; Vb[0]=(const float*)d_in[55]; Vg[0]=(const float*)d_in[56]; Vbe[0]=(const float*)d_in[57];
  Vw[1]=(const float*)d_in[58]; Vb[1]=(const float*)d_in[59]; Vg[1]=(const float*)d_in[60]; Vbe[1]=(const float*)d_in[61];
  Vw[2]=(const float*)d_in[62]; Vb[2]=(const float*)d_in[63]; Vg[2]=(const float*)d_in[64]; Vbe[2]=(const float*)d_in[65];
  Vw[3]=(const float*)d_in[66]; Vb[3]=(const float*)d_in[67]; Vg[3]=(const float*)d_in[68]; Vbe[3]=(const float*)d_in[69];
  Vw[4]=(const float*)d_in[70]; Vb[4]=(const float*)d_in[71];

  float* ws = (float*)d_ws;
  const size_t TOK = (size_t)MTOK * D_;     // 4,194,304 floats
  float* t   = ws;                          // tokens (residual stream)
  float* q   = ws + TOK;
  float* k   = ws + 2 * TOK;
  float* v   = ws + 3 * TOK;
  float* cx  = ws + 4 * TOK;                // attention context
  float* u   = ws + 5 * TOK;                // FF intermediate (8192 x 2048)
  float* rep = u;                           // head buffers reuse u after layers
  float* hA  = u + 65536;
  float* hB  = hA + 262144;

  // ----- board encoder: conv -> BN(eval) -> GELU, x4 -----
  conv_kernel<<<B_ * 128, 64, 0, stream>>>(x, cw[0], cb[0], cg[0], cbe[0], q, 19, 128, 3);
  conv_kernel<<<B_ * 256, 64, 0, stream>>>(q, cw[1], cb[1], cg[1], cbe[1], k, 128, 256, 3);
  conv_kernel<<<B_ * 256, 64, 0, stream>>>(k, cw[2], cb[2], cg[2], cbe[2], q, 256, 256, 3);
  conv_kernel<<<B_ * 512, 64, 0, stream>>>(q, cw[3], cb[3], cg[3], cbe[3], k, 256, 512, 1);
  tokens_kernel<<<(MTOK * D_) / 256, 256, 0, stream>>>(k, pe, lpe, t);

  // ----- 12 transformer layers -----
  for (int l = 0; l < L_; ++l) {
    const size_t oD2 = (size_t)l * D_ * D_;
    const size_t oD  = (size_t)l * D_;
    launch_gemm(t, Wq + oD2, bq + oD, q, MTOK, D_, D_, stream);
    launch_gemm(t, Wk + oD2, bk + oD, k, MTOK, D_, D_, stream);
    launch_gemm(t, Wv + oD2, bv + oD, v, MTOK, D_, D_, stream);
    attention_kernel<<<B_ * H_, 256, 0, stream>>>(q, k, v, relW + (size_t)l * 127 * HD_, cx);
    launch_gemm(cx, Wo + oD2, bo + oD, q, MTOK, D_, D_, stream);
    add_ln_kernel<<<MTOK, 256, 0, stream>>>(t, q, ln1g + oD, ln1b + oD);
    launch_gemm(t, W1 + (size_t)l * D_ * FF_, b1 + (size_t)l * FF_, u, MTOK, FF_, D_, stream);
    ln_act_kernel<<<MTOK, 256, 0, stream>>>(u, lnfg + (size_t)l * FF_, lnfb + (size_t)l * FF_, FF_, 1);
    launch_gemm(u, W2 + (size_t)l * FF_ * D_, b2 + oD, q, MTOK, D_, FF_, stream);
    add_ln_kernel<<<MTOK, 256, 0, stream>>>(t, q, ln2g + oD, ln2b + oD);
  }

  // ----- final LN + mean pool -----
  add_ln_kernel<<<MTOK, 256, 0, stream>>>(t, nullptr, nfg, nfb);
  mean_pool_kernel<<<B_, 512, 0, stream>>>(t, rep);

  float* logits = (float*)d_out;
  float* valout = (float*)d_out + (size_t)B_ * 4672;

  // ----- policy head -----
  launch_gemm(rep, Pw[0], Pb[0], hA, B_, 1024, 512, stream);
  ln_act_kernel<<<B_, 256, 0, stream>>>(hA, Pg[0], Pbe[0], 1024, 1);
  launch_gemm(hA, Pw[1], Pb[1], hB, B_, 1024, 1024, stream);
  ln_act_kernel<<<B_, 256, 0, stream>>>(hB, Pg[1], Pbe[1], 1024, 1);
  launch_gemm(hB, Pw[2], Pb[2], hA, B_, 2048, 1024, stream);
  ln_act_kernel<<<B_, 256, 0, stream>>>(hA, Pg[2], Pbe[2], 2048, 1);
  launch_gemm(hA, Pw[3], Pb[3], logits, B_, 4672, 2048, stream);   // N=4672 -> 64-wide tiles

  // ----- value head -----
  launch_gemm(rep, Vw[0], Vb[0], hB, B_, 512, 512, stream);
  ln_act_kernel<<<B_, 256, 0, stream>>>(hB, Vg[0], Vbe[0], 512, 1);
  launch_gemm(hB, Vw[1], Vb[1], hA, B_, 512, 512, stream);
  ln_act_kernel<<<B_, 256, 0, stream>>>(hA, Vg[1], Vbe[1], 512, 1);
  launch_gemm(hA, Vw[2], Vb[2], hB, B_, 256, 512, stream);
  ln_act_kernel<<<B_, 256, 0, stream>>>(hB, Vg[2], Vbe[2], 256, 1);
  launch_gemm(hB, Vw[3], Vb[3], hA, B_, 128, 256, stream);
  ln_act_kernel<<<B_, 256, 0, stream>>>(hA, Vg[3], Vbe[3], 128, 1);
  value_final_kernel<<<1, 128, 0, stream>>>(hA, Vw[4], Vb[4], valout);
}
